// DotProductAttention_22883585753376
// MI455X (gfx1250) — matrix operand order repaired and checked by reading, not compiled, not run on hardware
//
#include <hip/hip_runtime.h>
#include <hip/hip_bf16.h>
#include <math.h>

typedef __attribute__((ext_vector_type(16))) _Float16 v16h;
typedef __attribute__((ext_vector_type(8)))  _Float16 v8h;
typedef __attribute__((ext_vector_type(8)))  float    v8f;
typedef __attribute__((ext_vector_type(2)))  __fp16   h2;
typedef __attribute__((ext_vector_type(4)))  __fp16   h4;
typedef __attribute__((ext_vector_type(8)))  __fp16   h8;

// Problem constants (from reference: B=2, H=16, S=2048, D=64)
static constexpr int kS = 2048;
static constexpr int kD = 64;

// ---- Fused DPP16 butterfly steps (one VALU op each). A DPP "row" = 16 lanes,
// exactly the WMMA C/D half-wave row groups.
#define DPP_MAX_STEP(v, CTRL)                                                  \
    {                                                                          \
        float _t;                                                              \
        asm("v_max_num_f32_dpp %0, %1, %1 " CTRL                               \
            " row_mask:0xf bank_mask:0xf bound_ctrl:1"                         \
            : "=v"(_t) : "v"(v));                                              \
        (v) = _t;                                                              \
    }
#define DPP_ADD_STEP(v, CTRL)                                                  \
    {                                                                          \
        float _t;                                                              \
        asm("v_add_f32_dpp %0, %1, %1 " CTRL                                   \
            " row_mask:0xf bank_mask:0xf bound_ctrl:1"                         \
            : "=v"(_t) : "v"(v));                                              \
        (v) = _t;                                                              \
    }
#define VMAX3(d, a, b, c)                                                      \
    asm("v_max3_num_f32 %0, %1, %2, %3" : "=v"(d) : "v"(a), "v"(b), "v"(c))

__device__ __forceinline__ float row16_max(float v) {
    DPP_MAX_STEP(v, "quad_perm:[1,0,3,2]");
    DPP_MAX_STEP(v, "quad_perm:[2,3,0,1]");
    DPP_MAX_STEP(v, "row_half_mirror");
    DPP_MAX_STEP(v, "row_mirror");
    return v;
}
__device__ __forceinline__ float row16_sum(float v) {
    DPP_ADD_STEP(v, "quad_perm:[1,0,3,2]");
    DPP_ADD_STEP(v, "quad_perm:[2,3,0,1]");
    DPP_ADD_STEP(v, "row_half_mirror");
    DPP_ADD_STEP(v, "row_mirror");
    return v;
}

// Staged chunk held in registers between global load and LDS store so the
// staging ds_stores issue AFTER the WMMA operand ds_loads (DScnt counts both).
struct Staged {
    h8 krow[4];  // 32 f16 of one K row portion
    h2 vt[16];   // 16 (key, key+1) packs of V^T
};

// Flash attention, wave32 / WMMA 16x16x32 f16, 64-key chunks, double-buffered
// LDS. Block = 128 threads = 4 waves; each wave owns a 16-query tile, the
// block shares one (b,h). Softmax in base-2 (log2e folded into Q scale).
// The softmax denominator is carried as per-lane partials (alpha is uniform
// within each 16-lane row group, so the linear recurrence commutes with the
// cross-lane sum) and reduced once in the epilogue.
__global__ __launch_bounds__(128)
void fa_wmma_kernel(const float* __restrict__ Q,
                    const float* __restrict__ K,
                    const float* __restrict__ V,
                    const int*   __restrict__ mask,
                    float*       __restrict__ O)
{
    __shared__ __align__(32) _Float16 ldsK [2][64 * kD];   // key-major K chunk
    __shared__ __align__(32) _Float16 ldsVT[2][kD * 64];   // d-major V^T chunk
    __shared__ __align__(32) _Float16 ldsP [4][16 * 64];   // per-wave P slice

    const int tid  = threadIdx.x;
    const int wave = tid >> 5;
    const int lane = tid & 31;
    const int hs   = lane >> 4;
    const int ln   = lane & 15;

    const int    bh    = blockIdx.y;
    const size_t base  = (size_t)bh * kS * kD;
    const int    qtile = blockIdx.x * 64 + wave * 16;

    // ---- Q tile (16 x 64) in WMMA A-layout registers; scale = log2e/sqrt(D).
    v16h aQ0, aQ1;
    {
        const float* qrow  = Q + base + (size_t)(qtile + ln) * kD;
        const float  scale = 0.125f * 1.44269504088896340736f;
        #pragma unroll
        for (int i = 0; i < 16; ++i) {
            const int dd = 16 * hs + i;
            aQ0[i] = (_Float16)(qrow[dd]      * scale);
            aQ1[i] = (_Float16)(qrow[32 + dd] * scale);
        }
    }

    float mrow[8], lpart[8];   // lpart: per-lane unreduced denominator partial
    v8f   acc[4];
    #pragma unroll
    for (int r = 0; r < 8; ++r) { mrow[r] = -INFINITY; lpart[r] = 0.0f; }
    #pragma unroll
    for (int t = 0; t < 4; ++t)
        #pragma unroll
        for (int r = 0; r < 8; ++r) acc[t][r] = 0.0f;

    // Cooperative loader coordinates.
    const int k_kk = tid >> 1;            // K: key (0..63)
    const int k_do = (tid & 1) * 32;      //    d offset (0 or 32)
    const int v_k  = (tid & 31) * 2;      // V^T: key pair base
    const int v_d  = (tid >> 5) * 16;     //      d group (0,16,32,48)

    // Global load + f32->f16 pack (v_cvt_pk_rtz_f16_f32, two floats per op).
    auto load_convert = [&](int kb) -> Staged {
        Staged st;
        const float* krow = K + base + (size_t)(kb + k_kk) * kD + k_do;
        #pragma unroll
        for (int g = 0; g < 4; ++g) {
            const h2 p0 = __builtin_amdgcn_cvt_pkrtz(krow[8 * g + 0], krow[8 * g + 1]);
            const h2 p1 = __builtin_amdgcn_cvt_pkrtz(krow[8 * g + 2], krow[8 * g + 3]);
            const h2 p2 = __builtin_amdgcn_cvt_pkrtz(krow[8 * g + 4], krow[8 * g + 5]);
            const h2 p3 = __builtin_amdgcn_cvt_pkrtz(krow[8 * g + 6], krow[8 * g + 7]);
            const h4 lo = __builtin_shufflevector(p0, p1, 0, 1, 2, 3);
            const h4 hi = __builtin_shufflevector(p2, p3, 0, 1, 2, 3);
            st.krow[g]  = __builtin_shufflevector(lo, hi, 0, 1, 2, 3, 4, 5, 6, 7);
        }
        const float* v0p = V + base + (size_t)(kb + v_k) * kD + v_d;
        const float* v1p = v0p + kD;
        #pragma unroll
        for (int i = 0; i < 16; ++i)
            st.vt[i] = __builtin_amdgcn_cvt_pkrtz(v0p[i], v1p[i]);
        return st;
    };
    auto store_staged = [&](const Staged& st, int b) {
        #pragma unroll
        for (int g = 0; g < 4; ++g)
            *(h8*)&ldsK[b][k_kk * kD + k_do + 8 * g] = st.krow[g];
        #pragma unroll
        for (int i = 0; i < 16; ++i)
            *(h2*)&ldsVT[b][(v_d + i) * 64 + v_k] = st.vt[i];
    };

    {
        Staged st0 = load_convert(0);
        store_staged(st0, 0);
    }

    int buf = 0;
    for (int kb = 0; kb < kS; kb += 64, buf ^= 1) {
        __syncthreads();  // buf ready; everyone done with buf^1

        // Kick next chunk's global loads early; LDS stores deferred to the end.
        Staged st;
        const bool have_next = (kb + 64 < kS);
        if (have_next) st = load_convert(kb + 64);

        // ---- Scores: four 16x16 key tiles, two K=32 WMMAs each (d = 64).
        const _Float16* lk = ldsK[buf];
        v8f s[4];
        #pragma unroll
        for (int j = 0; j < 4; ++j) {
            const v16h b0 = *(const v16h*)&lk[(16 * j + ln) * kD +      16 * hs];
            const v16h b1 = *(const v16h*)&lk[(16 * j + ln) * kD + 32 + 16 * hs];
            v8f sj = {};
            sj = __builtin_amdgcn_wmma_f32_16x16x32_f16(false, aQ0, false, b0, (short)0, sj, false, false);
            sj = __builtin_amdgcn_wmma_f32_16x16x32_f16(false, aQ1, false, b1, (short)0, sj, false, false);
            s[j] = sj;
        }

        // ---- Additive mask + online softmax (base-2), DPP16 max reduction.
        _Float16* P = ldsP[wave];
        float alpha[8];
        #pragma unroll
        for (int r = 0; r < 8; ++r) {
            const int  q  = qtile + r + 8 * hs;
            const int* mp = mask + (size_t)q * kS + kb;
            float sc[4];
            #pragma unroll
            for (int j = 0; j < 4; ++j)   // -1e9 * log2(e)
                sc[j] = s[j][r] - 1.442695e9f * (float)mp[16 * j + ln];

            // mrow[r] is uniform within the 16-lane group -> fold it in before
            // the cross-lane reduction.
            float vmax;
            VMAX3(vmax, sc[0], sc[1], sc[2]);
            VMAX3(vmax, vmax, sc[3], mrow[r]);
            const float mnew = row16_max(vmax);

            const float a = __builtin_amdgcn_exp2f(mrow[r] - mnew);  // 1st: 0
            const int row = r + 8 * hs;
            float psum = 0.0f;
            #pragma unroll
            for (int j = 0; j < 4; ++j) {
                const float p = __builtin_amdgcn_exp2f(sc[j] - mnew);
                psum += p;
                P[row * 64 + 16 * j + ln] = (_Float16)p;
            }
            // Keep denominator per-lane; reduce across lanes once at the end.
            lpart[r] = lpart[r] * a + psum;
            mrow[r]  = mnew;
            alpha[r] = a;
        }
        v8f av;
        #pragma unroll
        for (int r = 0; r < 8; ++r) av[r] = alpha[r];
        #pragma unroll
        for (int t = 0; t < 4; ++t) acc[t] *= av;   // v_pk_mul_f32 candidates

        // ---- P (16x64) back as two f16 A-layout operands.
        v16h aP0, aP1;
        {
            const v8h lo0 = *(const v8h*)&P[ln * 64 +      16 * hs];
            const v8h hi0 = *(const v8h*)&P[ln * 64 + 8  + 16 * hs];
            const v8h lo1 = *(const v8h*)&P[ln * 64 + 32 + 16 * hs];
            const v8h hi1 = *(const v8h*)&P[ln * 64 + 40 + 16 * hs];
            aP0 = __builtin_shufflevector(lo0, hi0, 0, 1, 2, 3, 4, 5, 6, 7,
                                                    8, 9, 10, 11, 12, 13, 14, 15);
            aP1 = __builtin_shufflevector(lo1, hi1, 0, 1, 2, 3, 4, 5, 6, 7,
                                                    8, 9, 10, 11, 12, 13, 14, 15);
        }

        // ---- O(16x64) += P(16x64) * V(64x64).
        const _Float16* lv = ldsVT[buf];
        #pragma unroll
        for (int t = 0; t < 4; ++t) {
            const v16h bv0 = *(const v16h*)&lv[(t * 16 + ln) * 64 +      16 * hs];
            const v16h bv1 = *(const v16h*)&lv[(t * 16 + ln) * 64 + 32 + 16 * hs];
            acc[t] = __builtin_amdgcn_wmma_f32_16x16x32_f16(false, aP0, false, bv0, (short)0, acc[t], false, false);
            acc[t] = __builtin_amdgcn_wmma_f32_16x16x32_f16(false, aP1, false, bv1, (short)0, acc[t], false, false);
        }

        // ---- Deferred staging stores (after all WMMA operand ds_loads).
        if (have_next) store_staged(st, buf ^ 1);
    }

    // ---- Epilogue: reduce denominator partials across the 16-lane groups,
    // normalize (rcp + vector mul), store fp32.
    v8f rv;
    #pragma unroll
    for (int r = 0; r < 8; ++r)
        rv[r] = __builtin_amdgcn_rcpf(row16_sum(lpart[r]));
    #pragma unroll
    for (int t = 0; t < 4; ++t) {
        acc[t] *= rv;
        #pragma unroll
        for (int r = 0; r < 8; ++r) {
            const int q = qtile + r + 8 * hs;
            O[base + (size_t)q * kD + t * 16 + ln] = acc[t][r];
        }
    }
}

extern "C" void kernel_launch(void* const* d_in, const int* in_sizes, int n_in,
                              void* d_out, int out_size, void* d_ws, size_t ws_size,
                              hipStream_t stream) {
    const float* Q    = (const float*)d_in[0];
    const float* K    = (const float*)d_in[1];
    const float* V    = (const float*)d_in[2];
    const int*   mask = (const int*)d_in[3];
    // d_in[4] = key_dim (64), folded into the kernel as a constant.
    float* O = (float*)d_out;

    const int BH = in_sizes[0] / (kS * kD);  // batch * heads (= 32)
    dim3 grid(kS / 64, BH);
    dim3 block(128);
    fa_wmma_kernel<<<grid, block, 0, stream>>>(Q, K, V, mask, O);
}
